// MultiheadSpatialAttn_59416577573288
// MI455X (gfx1250) — matrix operand order repaired and checked by reading, not compiled, not run on hardware
//
#include <hip/hip_runtime.h>
#include <hip/hip_bf16.h>

// ---------------------------------------------------------------------------
// Types for gfx1250 wave32 WMMA (V_WMMA_F32_16X16X32_BF16)
// ---------------------------------------------------------------------------
typedef __attribute__((ext_vector_type(16))) __bf16          bf16x16;
typedef __attribute__((ext_vector_type(8)))  float           f32x8;
typedef __attribute__((ext_vector_type(8)))  unsigned short  u16x8;
typedef __attribute__((ext_vector_type(16))) unsigned short  u16x16;

#define B_    8
#define SQ_   1024
#define SKV_  1024
#define D_    256
#define H_    8
#define DH_   32

__device__ __forceinline__ unsigned short f32_to_bf16_bits(float f) {
    unsigned int u = __float_as_uint(f);
    unsigned int r = u + 0x7fffu + ((u >> 16) & 1u);   // round-to-nearest-even
    return (unsigned short)(r >> 16);
}

__device__ __forceinline__ bf16x16 make_frag(u16x8 lo, u16x8 hi) {
    u16x16 t;
#pragma unroll
    for (int i = 0; i < 8; ++i) { t[i] = lo[i]; t[i + 8] = hi[i]; }
    return __builtin_bit_cast(bf16x16, t);
}

// A fragment: 16(M) x 32(K), row-major source, ld in elements.
// Lane L holds row (L&15); elems 0..7 -> K=(L>>4)*8+j ; 8..15 -> K=16+(L>>4)*8+(j-8)
__device__ __forceinline__ bf16x16 load_a_frag(const unsigned short* src, int ld, int lane) {
    int row  = lane & 15;
    int koff = (lane >> 4) * 8;
    const unsigned short* p = src + row * ld;
    u16x8 lo = *(const u16x8*)(p + koff);
    u16x8 hi = *(const u16x8*)(p + 16 + koff);
    return make_frag(lo, hi);
}

// B fragment: 32(K) x 16(N). src + n*ld is 32 contiguous K elements (B-column).
// Lane L holds col (L&15); elems 0..7 -> K=(L>>4)*8+j ; 8..15 -> K=16+(L>>4)*8+(j-8)
__device__ __forceinline__ bf16x16 load_b_frag(const unsigned short* src, int ld, int lane) {
    int n    = lane & 15;
    int koff = (lane >> 4) * 8;
    const unsigned short* p = src + n * ld + koff;
    u16x8 lo = *(const u16x8*)(p);
    u16x8 hi = *(const u16x8*)(p + 16);
    return make_frag(lo, hi);
}

__device__ __forceinline__ f32x8 wmma_bf16(bf16x16 a, bf16x16 b, f32x8 c) {
    return __builtin_amdgcn_wmma_f32_16x16x32_bf16(false, a, false, b, (short)0, c,
                                                   false, false);
}

// CDNA5 async global->LDS copy (VFLAT async, tracked by ASYNCcnt).
// vdst = VGPR holding LDS byte offset; vaddr = 64-bit global address.
__device__ __forceinline__ void async_copy_b128(const unsigned short* g, unsigned lds_off) {
    asm volatile("global_load_async_to_lds_b128 %0, %1, off"
                 :
                 : "v"(lds_off), "v"(g)
                 : "memory");
}

__device__ __forceinline__ void wait_async0() {
    asm volatile("s_wait_asynccnt 0x0" ::: "memory");
}

// ---------------------------------------------------------------------------
// f32 -> bf16 elementwise conversion
// ---------------------------------------------------------------------------
__global__ void cvt_bf16_kernel(const float* __restrict__ src,
                                unsigned short* __restrict__ dst, int n) {
    int i = blockIdx.x * blockDim.x + threadIdx.x;
    if (i < n) dst[i] = f32_to_bf16_bits(src[i]);
}

// wt[n*256 + k] = bf16(w[k*256 + n])  (transpose 256x256 weight, f32 -> bf16)
__global__ void wt_bf16_kernel(const float* __restrict__ w,
                               unsigned short* __restrict__ wt) {
    int n = blockIdx.x;
    int k = threadIdx.x;
    wt[n * D_ + k] = f32_to_bf16_bits(w[k * D_ + n]);
}

// ---------------------------------------------------------------------------
// Projection GEMM:  out[M,256] = A[M,256] @ W[256,256] + bias
//   mode 0: bf16 row-major out; mode 1: bf16 transposed-V out [B*H,32,SKV];
//   mode 2: f32 row-major out.  One wave = one 16x16 tile, 8 chained WMMAs.
// ---------------------------------------------------------------------------
__global__ void proj_gemm_kernel(const unsigned short* __restrict__ A,
                                 const unsigned short* __restrict__ Wt,
                                 const float* __restrict__ bias,
                                 void* __restrict__ out, int M, int mode) {
    int wid  = (blockIdx.x * blockDim.x + threadIdx.x) >> 5;
    int lane = threadIdx.x & 31;
    const int ntn = D_ / 16;
    int mt = wid / ntn;
    int nt = wid % ntn;
    int m0 = mt * 16;
    if (m0 >= M) return;
    int n0 = nt * 16;

    f32x8 acc = {};
#pragma unroll
    for (int k0 = 0; k0 < D_; k0 += 32) {
        bf16x16 a = load_a_frag(A + (size_t)m0 * D_ + k0, D_, lane);
        bf16x16 b = load_b_frag(Wt + (size_t)n0 * D_ + k0, D_, lane);
        acc = wmma_bf16(a, b, acc);
    }

    int   nl   = lane & 15;
    int   hv   = lane >> 4;
    int   col  = n0 + nl;
    float bvv  = bias[col];
#pragma unroll
    for (int r = 0; r < 8; ++r) acc[r] += bvv;

    if (mode == 0) {
        unsigned short* O = (unsigned short*)out;
#pragma unroll
        for (int r = 0; r < 8; ++r) {
            int row = m0 + r + hv * 8;
            O[(size_t)row * D_ + col] = f32_to_bf16_bits(acc[r]);
        }
    } else if (mode == 1) {
        unsigned short* O = (unsigned short*)out;
        int h = col >> 5;
        int d = col & 31;
#pragma unroll
        for (int r = 0; r < 8; ++r) {
            int row = m0 + r + hv * 8;
            int bb  = row >> 10;
            int s   = row & (SKV_ - 1);
            O[(((size_t)bb * H_ + h) * DH_ + d) * SKV_ + s] = f32_to_bf16_bits(acc[r]);
        }
    } else {
        float* O = (float*)out;
#pragma unroll
        for (int r = 0; r < 8; ++r) {
            int row = m0 + r + hv * 8;
            O[(size_t)row * D_ + col] = acc[r];
        }
    }
}

// ---------------------------------------------------------------------------
// Fused flash attention. Block = 8 waves, all on the SAME (b,h) (8 q-tiles).
// K/V 32-kv chunks are staged once per block into LDS via double-buffered
// global_load_async_to_lds_b128 (one b128 per thread per chunk), overlapped
// with WMMA + online softmax; s_wait_asynccnt + barrier closes each stage.
// Steady-state global traffic = modulator/mask stream only (nontemporal).
// ---------------------------------------------------------------------------
__global__ void attn_kernel(const unsigned short* __restrict__ Q,
                            const unsigned short* __restrict__ K,
                            const unsigned short* __restrict__ VT,
                            const unsigned char* __restrict__ mask,
                            const float* __restrict__ mod,
                            unsigned short* __restrict__ X) {
    __shared__ __align__(16) unsigned short ldsK[2][32 * 32];   // [kv][dh]
    __shared__ __align__(16) unsigned short ldsV[2][32 * 32];   // [dh][kv]
    __shared__ __align__(16) unsigned short pbuf_all[8][16 * 32];

    int wid  = (blockIdx.x * blockDim.x + threadIdx.x) >> 5;
    int lane = threadIdx.x & 31;
    int wib  = threadIdx.x >> 5;

    int qt = wid & (SQ_ / 16 - 1);   // 64 q tiles per (b,h)
    int bh = wid >> 6;               // identical for all waves of the block
    int b  = bh >> 3;
    int h  = bh & 7;
    int q0 = qt * 16;

    unsigned short* pbuf = pbuf_all[wib];
    const unsigned  kvbytes  = 32 * 32 * 2;                       // 2 KB / buffer
    const unsigned  ldsK_off = (unsigned)(uintptr_t)&ldsK[0][0];  // LDS byte offset
    const unsigned  ldsV_off = (unsigned)(uintptr_t)&ldsV[0][0];

    const unsigned short* kgbase  = K  + ((size_t)b  * SKV_) * D_ + h * DH_;
    const unsigned short* vtgbase = VT + ((size_t)bh * DH_) * SKV_;

    int nl = lane & 15;
    int hv = lane >> 4;

    // Q A-fragment (persistent across the whole kv loop)
    const unsigned short* qbase = Q + ((size_t)b * SQ_ + q0) * D_ + h * DH_;
    bf16x16 qfrag = load_a_frag(qbase, D_, lane);

    const float scale = 0.17677669529663687f;   // 32^-0.5

    f32x8 o0 = {}, o1 = {};
    float mrow[8], lrow[8];
#pragma unroll
    for (int r = 0; r < 8; ++r) { mrow[r] = -1e30f; lrow[r] = 0.0f; }

    const float*         modbase  = mod  + ((size_t)bh * SQ_ + q0) * SKV_;
    const unsigned char* maskbase = mask + ((size_t)b  * SQ_ + q0) * SKV_;

    // ---- stage helper: one b128 async load per thread (wave-aligned split) --
    auto stage = [&](int buf, int kv) {
        int t = threadIdx.x;
        if (t < 128) {                         // waves 0..3: K tile (32 kv x 32 dh)
            int row = t >> 2, c8 = (t & 3) * 8;
            async_copy_b128(kgbase + (size_t)(kv + row) * D_ + c8,
                            ldsK_off + (unsigned)buf * kvbytes +
                                (unsigned)(row * 32 + c8) * 2u);
        } else {                               // waves 4..7: V tile (32 dh x 32 kv)
            int c = t - 128;
            int d = c >> 2, c8 = (c & 3) * 8;
            async_copy_b128(vtgbase + (size_t)d * SKV_ + kv + c8,
                            ldsV_off + (unsigned)buf * kvbytes +
                                (unsigned)(d * 32 + c8) * 2u);
        }
    };

    // prologue: fill buffer 0 with kv chunk 0
    stage(0, 0);
    wait_async0();
    __syncthreads();

#pragma unroll 1
    for (int it = 0; it < SKV_ / 32; ++it) {
        int cur = it & 1;
        int kv0 = it * 32;
        if (it + 1 < SKV_ / 32) stage(1 - cur, kv0 + 32);   // async prefetch

        // ---- two 16x16 score tiles from LDS K ----
        f32x8 s[2];
#pragma unroll
        for (int t = 0; t < 2; ++t) {
            bf16x16 kfrag = load_b_frag(&ldsK[cur][t * 16 * 32], 32, lane);
            f32x8 z = {};
            s[t] = wmma_bf16(qfrag, kfrag, z);

            int col = kv0 + t * 16 + nl;
#pragma unroll
            for (int r = 0; r < 8; ++r) {
                int   row = r + hv * 8;
                float mv  = __builtin_nontemporal_load(
                                modbase + (size_t)row * SKV_ + col);
                unsigned char mb = __builtin_nontemporal_load(
                                maskbase + (size_t)row * SKV_ + col);
                float sv = s[t][r] * scale * mv;
                s[t][r]  = mb ? -1e30f : sv;
            }
        }

        // ---- online softmax (rows live across 16-lane half-waves) ----
#pragma unroll
        for (int r = 0; r < 8; ++r) {
            float mx = fmaxf(s[0][r], s[1][r]);
            mx = fmaxf(mx, __shfl_xor(mx, 1, 16));
            mx = fmaxf(mx, __shfl_xor(mx, 2, 16));
            mx = fmaxf(mx, __shfl_xor(mx, 4, 16));
            mx = fmaxf(mx, __shfl_xor(mx, 8, 16));
            float m_new = fmaxf(mrow[r], mx);

            float p0 = __expf(s[0][r] - m_new);
            float p1 = __expf(s[1][r] - m_new);
            s[0][r] = p0;
            s[1][r] = p1;

            float ps = p0 + p1;
            ps += __shfl_xor(ps, 1, 16);
            ps += __shfl_xor(ps, 2, 16);
            ps += __shfl_xor(ps, 4, 16);
            ps += __shfl_xor(ps, 8, 16);

            float alpha = __expf(mrow[r] - m_new);
            lrow[r] = lrow[r] * alpha + ps;
            mrow[r] = m_new;
            o0[r] *= alpha;
            o1[r] *= alpha;
        }

        // ---- stage P tile (C-layout -> row-major 16x32 bf16) in LDS ----
#pragma unroll
        for (int r = 0; r < 8; ++r) {
            int row = r + hv * 8;
            pbuf[row * 32 + nl]      = f32_to_bf16_bits(s[0][r]);
            pbuf[row * 32 + 16 + nl] = f32_to_bf16_bits(s[1][r]);
        }

        // ---- P @ V from LDS ----
        bf16x16 pfrag = load_a_frag(pbuf, 32, lane);
        bf16x16 vf0   = load_b_frag(&ldsV[cur][0],       32, lane);
        bf16x16 vf1   = load_b_frag(&ldsV[cur][16 * 32], 32, lane);
        o0 = wmma_bf16(pfrag, vf0, o0);
        o1 = wmma_bf16(pfrag, vf1, o1);

        // close this stage: our async issues complete, then block-wide rendezvous
        wait_async0();
        __syncthreads();
    }

    // ---- normalize and store x (bf16, row-major [B*SQ, 256]) ----
#pragma unroll
    for (int r = 0; r < 8; ++r) {
        float inv = 1.0f / lrow[r];
        int   row = q0 + r + hv * 8;
        unsigned short* xb = X + ((size_t)b * SQ_ + row) * D_ + h * DH_;
        xb[nl]      = f32_to_bf16_bits(o0[r] * inv);
        xb[16 + nl] = f32_to_bf16_bits(o1[r] * inv);
    }
}

// ---------------------------------------------------------------------------
// Host-side launch
// ---------------------------------------------------------------------------
extern "C" void kernel_launch(void* const* d_in, const int* in_sizes, int n_in,
                              void* d_out, int out_size, void* d_ws, size_t ws_size,
                              hipStream_t stream) {
    (void)in_sizes; (void)n_in; (void)out_size; (void)ws_size;

    const float*         query  = (const float*)d_in[0];
    const float*         keyval = (const float*)d_in[1];
    const unsigned char* mask   = (const unsigned char*)d_in[2];
    const float*         mod    = (const float*)d_in[3];
    const float*         wq     = (const float*)d_in[4];
    const float*         wk     = (const float*)d_in[5];
    const float*         wv     = (const float*)d_in[6];
    const float*         bq     = (const float*)d_in[7];
    const float*         bk     = (const float*)d_in[8];
    const float*         bv     = (const float*)d_in[9];
    const float*         ow     = (const float*)d_in[10];
    const float*         ob     = (const float*)d_in[11];

    const int M   = B_ * SQ_;          // 8192 rows for all GEMMs
    const int NEL = M * D_;            // 2,097,152 elements

    char*  ws  = (char*)d_ws;
    size_t off = 0;
    auto wsalloc = [&](size_t bytes) -> void* {
        void* p = ws + off;
        off += (bytes + 255) & ~(size_t)255;
        return p;
    };

    unsigned short* query_bf = (unsigned short*)wsalloc((size_t)NEL * 2);
    unsigned short* keyval_bf= (unsigned short*)wsalloc((size_t)NEL * 2);
    unsigned short* wqt      = (unsigned short*)wsalloc((size_t)D_ * D_ * 2);
    unsigned short* wkt      = (unsigned short*)wsalloc((size_t)D_ * D_ * 2);
    unsigned short* wvt      = (unsigned short*)wsalloc((size_t)D_ * D_ * 2);
    unsigned short* owt      = (unsigned short*)wsalloc((size_t)D_ * D_ * 2);
    unsigned short* qbf      = (unsigned short*)wsalloc((size_t)NEL * 2);
    unsigned short* kbf      = (unsigned short*)wsalloc((size_t)NEL * 2);
    unsigned short* vtbf     = (unsigned short*)wsalloc((size_t)NEL * 2);
    unsigned short* xbf      = (unsigned short*)wsalloc((size_t)NEL * 2);

    // 1) f32 -> bf16 activations, transposed bf16 weights
    cvt_bf16_kernel<<<NEL / 256, 256, 0, stream>>>(query,  query_bf,  NEL);
    cvt_bf16_kernel<<<NEL / 256, 256, 0, stream>>>(keyval, keyval_bf, NEL);
    wt_bf16_kernel<<<D_, D_, 0, stream>>>(wq, wqt);
    wt_bf16_kernel<<<D_, D_, 0, stream>>>(wk, wkt);
    wt_bf16_kernel<<<D_, D_, 0, stream>>>(wv, wvt);
    wt_bf16_kernel<<<D_, D_, 0, stream>>>(ow, owt);

    // 2) projections (one wave per 16x16 tile -> M waves -> M/8 blocks of 256)
    const int gemm_blocks = M / 8;     // 1024
    proj_gemm_kernel<<<gemm_blocks, 256, 0, stream>>>(query_bf,  wqt, bq, qbf,  M, 0);
    proj_gemm_kernel<<<gemm_blocks, 256, 0, stream>>>(keyval_bf, wkt, bk, kbf,  M, 0);
    proj_gemm_kernel<<<gemm_blocks, 256, 0, stream>>>(keyval_bf, wvt, bv, vtbf, M, 1);

    // 3) fused attention: B*H*(SQ/16) = 4096 waves -> 512 blocks of 256
    attn_kernel<<<(B_ * H_ * (SQ_ / 16)) / 8, 256, 0, stream>>>(
        qbf, kbf, vtbf, mask, mod, xbf);

    // 4) output projection -> f32 d_out
    proj_gemm_kernel<<<gemm_blocks, 256, 0, stream>>>(xbf, owt, ob, d_out, M, 2);
}
